// RNNModel_40106404610783
// MI455X (gfx1250) — matrix operand order repaired and checked by reading, not compiled, not run on hardware
//
#include <hip/hip_runtime.h>

#define BATCH 256
#define NATOM 8
#define TSTEPS 128
#define HID 128
#define ROWS_PER_WG 16
#define HSTRIDE (ROWS_PER_WG * HID)         /* elements per h buffer */
#define NWG (NATOM * (BATCH / ROWS_PER_WG)) /* 128 */
#define NTHREADS 256                        /* 8 waves (wave32) */

typedef __attribute__((ext_vector_type(16))) __bf16 bf16x16;
typedef __attribute__((ext_vector_type(8)))  float  f32x8;

union Frag {
    bf16x16 v;
    uint4   q[2];
    __bf16  h[16];
};

__device__ __forceinline__ __bf16 f2bf(float x) { return (__bf16)x; }
__device__ __forceinline__ float  bf2f(__bf16 x) { return (float)x; }

__device__ __forceinline__ float fsig(float x) {
    return __builtin_amdgcn_rcpf(1.0f + __expf(-x));
}
__device__ __forceinline__ float ftanh(float x) {
    float e = __expf(-2.0f * x);
    return (1.0f - e) * __builtin_amdgcn_rcpf(1.0f + e);
}

__device__ __forceinline__ f32x8 wmma_bf16(bf16x16 a, bf16x16 b, f32x8 c) {
    // v_wmma_f32_16x16x32_bf16
    return __builtin_amdgcn_wmma_f32_16x16x32_bf16(false, a, false, b,
                                                   (short)0, c, false, false);
}

// A-matrix fragment (16x32 bf16) from LDS, rows row-major, stride HID halves.
// lanes 0-15: row=l, K 0-15; lanes 16-31: same rows, K 16-31.
__device__ __forceinline__ Frag lds_fragA(const __bf16* base, int lane, int kstep) {
    const int l  = lane & 15;
    const int hi = lane >> 4;
    const __bf16* p = base + l * HID + kstep * 32 + hi * 16;
    Frag f;
    f.q[0] = *reinterpret_cast<const uint4*>(p);      // ds_load_b128
    f.q[1] = *reinterpret_cast<const uint4*>(p + 8);  // ds_load_b128
    return f;
}

// B-matrix fragment (32x16 bf16) from a global fp32 weight row [n][k].
// lanes 0-15: N=l, K 0-15; lanes 16-31: N=l, K 16-31.
__device__ __forceinline__ Frag gbl_fragB_f32(const float* rowPtr, int lane, int kstep) {
    const int hi = lane >> 4;
    const int kk = kstep * 32 + hi * 16;
    float4 a0 = *reinterpret_cast<const float4*>(rowPtr + kk);
    float4 a1 = *reinterpret_cast<const float4*>(rowPtr + kk + 4);
    float4 a2 = *reinterpret_cast<const float4*>(rowPtr + kk + 8);
    float4 a3 = *reinterpret_cast<const float4*>(rowPtr + kk + 12);
    float xs[16] = {a0.x, a0.y, a0.z, a0.w, a1.x, a1.y, a1.z, a1.w,
                    a2.x, a2.y, a2.z, a2.w, a3.x, a3.y, a3.z, a3.w};
    Frag f;
#pragma unroll
    for (int m = 0; m < 16; ++m) f.h[m] = f2bf(xs[m]);
    return f;
}

extern "C" __global__ __launch_bounds__(NTHREADS) void
lstm_wmma_kernel(const float* __restrict__ inputs,  // [B,A,T,H]
                 const float* __restrict__ w_ih,    // [A,4H,H]
                 const float* __restrict__ w_hh,    // [A,4H,H]
                 const float* __restrict__ b_ih,    // [A,4H]
                 const float* __restrict__ b_hh,    // [A,4H]
                 const int*   __restrict__ psPtr,   // prediction_steps
                 float* __restrict__ out)           // [B,A,T-1,H]
{
    // Double-buffered hidden state only; weights live in registers.
    extern __shared__ __bf16 smem[];                 // 2 x [16][128] bf16

    const int wg      = blockIdx.x;
    const int atom    = wg >> 4;
    const int rowBase = (wg & 15) * ROWS_PER_WG;
    const int tid     = threadIdx.x;
    const int lane    = tid & 31;
    const int wv      = tid >> 5;   // hidden-column tile j = 0..7
    const int l15     = lane & 15;
    const int hi16    = lane >> 4;
    const int ps      = psPtr[0];

    const float* wiA = w_ih + (size_t)atom * 512 * HID;
    const float* whA = w_hh + (size_t)atom * 512 * HID;

    // ---- Register-resident combined weights Wc = wi + wh, bf16 B-fragments.
    // bw[k][g]: 16 fragments x 8 VGPRs = 128 VGPRs per lane, loop-invariant.
    bf16x16 bw[4][4];
#pragma unroll
    for (int k = 0; k < 4; ++k) {
#pragma unroll
        for (int g = 0; g < 4; ++g) {
            const int n  = g * HID + wv * 16 + l15;
            const int kk = k * 32 + hi16 * 16;
            const float* wip = wiA + (size_t)n * HID + kk;
            const float* whp = whA + (size_t)n * HID + kk;
            float4 wa0 = *reinterpret_cast<const float4*>(wip);
            float4 wa1 = *reinterpret_cast<const float4*>(wip + 4);
            float4 wa2 = *reinterpret_cast<const float4*>(wip + 8);
            float4 wa3 = *reinterpret_cast<const float4*>(wip + 12);
            float4 wb0 = *reinterpret_cast<const float4*>(whp);
            float4 wb1 = *reinterpret_cast<const float4*>(whp + 4);
            float4 wb2 = *reinterpret_cast<const float4*>(whp + 8);
            float4 wb3 = *reinterpret_cast<const float4*>(whp + 12);
            float s[16] = {wa0.x + wb0.x, wa0.y + wb0.y, wa0.z + wb0.z, wa0.w + wb0.w,
                           wa1.x + wb1.x, wa1.y + wb1.y, wa1.z + wb1.z, wa1.w + wb1.w,
                           wa2.x + wb2.x, wa2.y + wb2.y, wa2.z + wb2.z, wa2.w + wb2.w,
                           wa3.x + wb3.x, wa3.y + wb3.y, wa3.z + wb3.z, wa3.w + wb3.w};
            Frag f;
#pragma unroll
            for (int m = 0; m < 16; ++m) f.h[m] = f2bf(s[m]);
            bw[k][g] = f.v;
        }
    }

    // Zero h buffer 0.
    for (int idx = tid; idx < HSTRIDE; idx += NTHREADS)
        smem[idx] = (__bf16)0.0f;
    __syncthreads();

    // Per-lane bias for the four gates at hidden column `col`.
    const int col = wv * 16 + l15;
    float bias[4];
#pragma unroll
    for (int g = 0; g < 4; ++g)
        bias[g] = b_ih[atom * 512 + g * HID + col] + b_hh[atom * 512 + g * HID + col];

    // Cell state in registers, WMMA accumulator layout.
    float cst[8];
#pragma unroll
    for (int p = 0; p < 8; ++p) cst[p] = 0.0f;

    const f32x8 vzero = {0.f, 0.f, 0.f, 0.f, 0.f, 0.f, 0.f, 0.f};

#pragma unroll 1
    for (int t = 0; t < TSTEPS - 1; ++t) {
        const int coff = (t & 1) * HSTRIDE;
        const __bf16* hc = smem + coff;               // read h(t)
        __bf16*       hn = smem + (HSTRIDE - coff);   // write h(t+1)

        f32x8 acc[4];
#pragma unroll
        for (int g = 0; g < 4; ++g) acc[g] = vzero;

        // g += h @ Wc^T  (covers both GEMMs on non-TF steps since x == h)
#pragma unroll
        for (int k = 0; k < 4; ++k) {
            Frag a = lds_fragA(hc, lane, k);
#pragma unroll
            for (int g = 0; g < 4; ++g)
                acc[g] = wmma_bf16(a.v, bw[k][g], acc[g]);
        }

        // Teacher-forcing correction: g += (x_t - h) @ wi^T  (13 of 127 steps)
        if ((t % ps) == 0) {
            if (t + ps < TSTEPS - 1) {
                const int bpre = rowBase + l15;
                __builtin_prefetch(
                    inputs + (((size_t)bpre * NATOM + atom) * TSTEPS + (t + ps)) * HID,
                    0, 0);
            }
#pragma unroll
            for (int k = 0; k < 4; ++k) {
                Frag hf = lds_fragA(hc, lane, k);
                const int brow = rowBase + l15;
                const float* xp =
                    inputs + (((size_t)brow * NATOM + atom) * TSTEPS + t) * HID;
                const int kk = k * 32 + hi16 * 16;
                float4 x0 = *reinterpret_cast<const float4*>(xp + kk);
                float4 x1 = *reinterpret_cast<const float4*>(xp + kk + 4);
                float4 x2 = *reinterpret_cast<const float4*>(xp + kk + 8);
                float4 x3 = *reinterpret_cast<const float4*>(xp + kk + 12);
                float xs[16] = {x0.x, x0.y, x0.z, x0.w, x1.x, x1.y, x1.z, x1.w,
                                x2.x, x2.y, x2.z, x2.w, x3.x, x3.y, x3.z, x3.w};
                Frag d;
#pragma unroll
                for (int m = 0; m < 16; ++m)
                    d.h[m] = f2bf(xs[m] - bf2f(hf.h[m]));
#pragma unroll
                for (int g = 0; g < 4; ++g) {
                    const int n = g * HID + wv * 16 + l15;
                    Frag bwi = gbl_fragB_f32(wiA + (size_t)n * HID, lane, k);
                    acc[g] = wmma_bf16(d.v, bwi.v, acc[g]);
                }
            }
        }

        // Pointwise LSTM update: i/f/g/o for a given (b,h) share lane+slot.
#pragma unroll
        for (int p = 0; p < 8; ++p) {
            float gi = acc[0][p] + bias[0];
            float gf = acc[1][p] + bias[1];
            float gg = acc[2][p] + bias[2];
            float go = acc[3][p] + bias[3];
            float c  = fsig(gf) * cst[p] + fsig(gi) * ftanh(gg);
            cst[p]   = c;
            float h  = fsig(go) * ftanh(c);
            const int rowInBlk = p + hi16 * 8;
            hn[rowInBlk * HID + col] = f2bf(h);
            out[(((size_t)(rowBase + rowInBlk) * NATOM + atom) * (TSTEPS - 1) + t)
                    * HID + col] = h;
        }

        __syncthreads();  // h(t+1) complete & visible; old buffer free for reuse
    }
}

extern "C" void kernel_launch(void* const* d_in, const int* in_sizes, int n_in,
                              void* d_out, int out_size, void* d_ws, size_t ws_size,
                              hipStream_t stream) {
    const float* inputs = (const float*)d_in[0];
    const float* w_ih   = (const float*)d_in[1];
    const float* w_hh   = (const float*)d_in[2];
    const float* b_ih   = (const float*)d_in[3];
    const float* b_hh   = (const float*)d_in[4];
    const int*   ps     = (const int*)d_in[5];
    float*       out    = (float*)d_out;
    (void)in_sizes; (void)n_in; (void)out_size; (void)d_ws; (void)ws_size;

    const size_t shmem = (size_t)2 * HSTRIDE * sizeof(__bf16); // 8 KB
    lstm_wmma_kernel<<<dim3(NWG), dim3(NTHREADS), shmem, stream>>>(
        inputs, w_ih, w_hh, b_ih, b_hh, ps, out);
}
